// FLASH_ShareA_FFConvM_87385404604856
// MI455X (gfx1250) — matrix operand order repaired and checked by reading, not compiled, not run on hardware
//
#include <hip/hip_runtime.h>
#include <hip/hip_bf16.h>
#include <cstdint>

// ---------------------------------------------------------------------------
// MossFormer FLASH_ShareA_FFConvM block for MI455X (gfx1250), bf16 WMMA path,
// async global->LDS staging (ASYNCcnt) with double-buffered GEMM tiles.
// ---------------------------------------------------------------------------

typedef unsigned short u16;
typedef __attribute__((ext_vector_type(16))) __bf16 v16bf;
typedef __attribute__((ext_vector_type(8)))  __bf16 v8bf;
typedef __attribute__((ext_vector_type(8)))  float  v8f;
typedef int v4i __attribute__((vector_size(16)));   // matches async-LDS builtin

#define NT   16384   // tokens
#define DIM  512     // model dim
#define H2   2048    // to_hidden dim (v|u concatenated)
#define HD   1024    // v / u dim
#define QKD  128     // qk dim
#define GSZ  256     // attention group size
#define KERN 17      // depthwise conv taps
#define KPAD 8       // (KERN-1)/2

#define BM 128
#define BN 64
#define BK 32

#if defined(__has_builtin)
#if __has_builtin(__builtin_amdgcn_global_load_async_to_lds_b128)
#define HAVE_ASYNC_LDS 1
#endif
#endif
#ifndef HAVE_ASYNC_LDS
#define HAVE_ASYNC_LDS 0
#endif

__device__ __forceinline__ u16 f2bf(float f) {
  unsigned u = __builtin_bit_cast(unsigned, f);
  unsigned r = (u + 0x7FFFu + ((u >> 16) & 1u)) >> 16;
  return (u16)r;
}
__device__ __forceinline__ float bf2f(u16 h) {
  unsigned u = ((unsigned)h) << 16;
  return __builtin_bit_cast(float, u);
}
__device__ __forceinline__ float siluf(float x) { return x / (1.f + __expf(-x)); }
__device__ __forceinline__ float sigmf(float x) { return 1.f / (1.f + __expf(-x)); }

// 16-byte global -> LDS copy: async (ASYNCcnt, memory->LDS direct) if the
// gfx1250 builtin is available, else classic load+ds_store.
__device__ __forceinline__ void copy16(const u16* g, u16* l) {
#if HAVE_ASYNC_LDS
  __builtin_amdgcn_global_load_async_to_lds_b128(
      (__attribute__((address_space(1))) v4i*)g,
      (__attribute__((address_space(3))) v4i*)l, 0, 0);
#else
  *(uint4*)l = *(const uint4*)g;
#endif
}

// Wait for this wave's async LDS writes, then workgroup barrier.
__device__ __forceinline__ void async_fence_barrier() {
#if HAVE_ASYNC_LDS
#if defined(__has_builtin) && __has_builtin(__builtin_amdgcn_s_wait_asynccnt)
  __builtin_amdgcn_s_wait_asynccnt(0);
#else
  asm volatile("s_wait_asynccnt 0" ::: "memory");
#endif
#endif
  __syncthreads();
}

// Build a 16-element bf16 WMMA fragment from two contiguous 8-element chunks.
__device__ __forceinline__ v16bf frag16(const u16* lo, const u16* hi) {
  v8bf l = *(const v8bf*)lo;
  v8bf h = *(const v8bf*)hi;
  v16bf r;
#pragma unroll
  for (int i = 0; i < 8; ++i) { r[i] = l[i]; r[8 + i] = h[i]; }
  return r;
}

// ---------------------------------------------------------------------------
// LayerNorm (optionally with MossFormer token-shift on first half of channels)
// ---------------------------------------------------------------------------
__global__ __launch_bounds__(256) void ln_kernel(const float* __restrict__ X,
    const float* __restrict__ g, const float* __restrict__ b,
    u16* __restrict__ Y, int C, int shift_half) {
  __shared__ float rs[256], rq[256];
  const int row = blockIdx.x, tid = threadIdx.x;
  const int per = C >> 8;           // 2 for C=512, 4 for C=1024
  float vals[4];
  float s = 0.f, q = 0.f;
  for (int i = 0; i < per; ++i) {
    int c = tid + (i << 8);
    float v;
    if (shift_half && c < (C >> 1))
      v = (row > 0) ? X[(size_t)(row - 1) * C + c] : 0.f;
    else
      v = X[(size_t)row * C + c];
    vals[i] = v; s += v; q += v * v;
  }
  rs[tid] = s; rq[tid] = q;
  __syncthreads();
  for (int st = 128; st > 0; st >>= 1) {
    if (tid < st) { rs[tid] += rs[tid + st]; rq[tid] += rq[tid + st]; }
    __syncthreads();
  }
  const float mu  = rs[0] / (float)C;
  const float var = rq[0] / (float)C - mu * mu;
  const float ri  = rsqrtf(var + 1e-5f);
  for (int i = 0; i < per; ++i) {
    int c = tid + (i << 8);
    Y[(size_t)row * C + c] = f2bf((vals[i] - mu) * ri * g[c] + b[c]);
  }
}

// ---------------------------------------------------------------------------
// Generic bf16 GEMM:  C[M,N] = act(A[M,K] @ Bt[N,K]^T + bias)
//   outmode 0: store bf16     outmode 1: C(f32) += result
// 128x64 block tile, 8 waves (4x2), 2x2 WMMA 16x16x32 bf16 per wave,
// double-buffered LDS with async global->LDS staging.
// ---------------------------------------------------------------------------
__global__ __launch_bounds__(256) void gemm_kernel(const u16* __restrict__ A,
    const u16* __restrict__ Bt, const float* __restrict__ bias,
    void* __restrict__ Cout, int M, int N, int K, int act, int outmode) {
  __shared__ __align__(16) u16 As[2][BM][BK + 8];
  __shared__ __align__(16) u16 Bs[2][BN][BK + 8];
  const int tid  = threadIdx.x;
  const int lane = tid & 31, wave = tid >> 5;
  const int wm = wave & 3, wn = wave >> 2;        // 4 x 2 wave grid
  const int bm = blockIdx.y * BM, bn = blockIdx.x * BN;
  const int lr = lane & 15, lh = lane >> 4;

  auto stage = [&](int b, int k0) {
#pragma unroll
    for (int i = 0; i < 2; ++i) {                 // 512 x 16B of A tile
      int j = tid + (i << 8);
      int r = j >> 2, cw = (j & 3) << 3;
      copy16(A + (size_t)(bm + r) * K + k0 + cw, &As[b][r][cw]);
    }
    {                                             // 256 x 16B of B tile
      int r = tid >> 2, cw = (tid & 3) << 3;
      copy16(Bt + (size_t)(bn + r) * K + k0 + cw, &Bs[b][r][cw]);
    }
  };

  v8f acc[2][2] = {};
  stage(0, 0);
  async_fence_barrier();
  int buf = 0;
  for (int k0 = 0; k0 < K; k0 += BK) {
    if (k0 + BK < K) stage(buf ^ 1, k0 + BK);     // overlap fetch with WMMA
    if (k0 + 2 * BK < K)
      __builtin_prefetch(A + (size_t)(bm + (tid >> 1)) * K + k0 + 2 * BK, 0, 1);
#pragma unroll
    for (int mi = 0; mi < 2; ++mi) {
      const u16* arp = &As[buf][wm * 32 + mi * 16 + lr][0];
      v16bf af = frag16(arp + lh * 8, arp + lh * 8 + 16);
#pragma unroll
      for (int ni = 0; ni < 2; ++ni) {
        const u16* brp = &Bs[buf][wn * 32 + ni * 16 + lr][0];
        v16bf bf = frag16(brp + lh * 8, brp + lh * 8 + 16);
        acc[mi][ni] = __builtin_amdgcn_wmma_f32_16x16x32_bf16(
            false, af, false, bf, (short)0, acc[mi][ni], false, false);
      }
    }
    async_fence_barrier();
    buf ^= 1;
  }

#pragma unroll
  for (int mi = 0; mi < 2; ++mi)
#pragma unroll
    for (int ni = 0; ni < 2; ++ni) {
      const int n  = bn + wn * 32 + ni * 16 + lr;
      const float bv = bias ? bias[n] : 0.f;
#pragma unroll
      for (int r = 0; r < 8; ++r) {
        const int m = bm + wm * 32 + mi * 16 + r + (lh << 3);
        float v = acc[mi][ni][r] + bv;
        if (act) v = siluf(v);
        if (outmode == 0) ((u16*)Cout)[(size_t)m * N + n] = f2bf(v);
        else ((float*)Cout)[(size_t)m * N + n] += v;
      }
    }
}

// ---------------------------------------------------------------------------
// Grouped quadratic attention: per (group, 64-row chunk):
//   attn = relu(qq @ kk^T / GS)^2  (bf16 in LDS)  then  attn @ V, attn @ U.
// ---------------------------------------------------------------------------
__global__ __launch_bounds__(256) void attn_kernel(const u16* __restrict__ Q,
    const u16* __restrict__ Km, const u16* __restrict__ V,
    const u16* __restrict__ U, float* __restrict__ AttV,
    float* __restrict__ AttU) {
  __shared__ __align__(16) u16 AttS[64][264];     // 64 x 256 attn tile (bf16)
  __shared__ __align__(16) u16 Scr[12800];        // phase1: Qs+Ks / phase2: Vs
  const int tid  = threadIdx.x;
  const int lane = tid & 31, wave = tid >> 5;
  const int lr = lane & 15, lh = lane >> 4;
  const int grp = blockIdx.x >> 2, rc = blockIdx.x & 3;
  const int base  = grp * GSZ;
  const int rbase = base + rc * 64;

  u16 (*Qs)[40] = (u16(*)[40])(&Scr[0]);          // 64 x 32 (+pad)
  u16 (*Ks)[40] = (u16(*)[40])(&Scr[2560]);       // 256 x 32 (+pad)

  { // ---- phase 1: sim = qq @ kk^T, scale, relu^2 -> AttS (bf16) ----
    const int wm = wave & 1, wn = wave >> 1;      // 2 x 4 wave grid (64x256)
    v8f acc[2][4] = {};
    for (int k0 = 0; k0 < QKD; k0 += BK) {
      {
        int r = tid >> 2, cw = (tid & 3) << 3;    // 256 x 16B of Q rows
        copy16(Q + (size_t)(rbase + r) * QKD + k0 + cw, &Qs[r][cw]);
      }
#pragma unroll
      for (int i = 0; i < 4; ++i) {               // 1024 x 16B of K rows
        int j = tid + (i << 8);
        int r = j >> 2, cw = (j & 3) << 3;
        copy16(Km + (size_t)(base + r) * QKD + k0 + cw, &Ks[r][cw]);
      }
      async_fence_barrier();
#pragma unroll
      for (int mi = 0; mi < 2; ++mi) {
        const u16* arp = &Qs[wm * 32 + mi * 16 + lr][0];
        v16bf af = frag16(arp + lh * 8, arp + lh * 8 + 16);
#pragma unroll
        for (int ni = 0; ni < 4; ++ni) {
          const u16* brp = &Ks[wn * 64 + ni * 16 + lr][0];
          v16bf bf = frag16(brp + lh * 8, brp + lh * 8 + 16);
          acc[mi][ni] = __builtin_amdgcn_wmma_f32_16x16x32_bf16(
              false, af, false, bf, (short)0, acc[mi][ni], false, false);
        }
      }
      __syncthreads();
    }
    const float sc = 1.f / (float)GSZ;
#pragma unroll
    for (int mi = 0; mi < 2; ++mi)
#pragma unroll
      for (int ni = 0; ni < 4; ++ni)
#pragma unroll
        for (int r = 0; r < 8; ++r) {
          int m = wm * 32 + mi * 16 + r + (lh << 3);
          int n = wn * 64 + ni * 16 + lr;
          float s = fmaxf(acc[mi][ni][r] * sc, 0.f);
          AttS[m][n] = f2bf(s * s);
        }
  }
  __syncthreads();

  // ---- phase 2: quad = attn @ V (cols 0..1023) and attn @ U ----
  u16 (*Vs)[264] = (u16(*)[264])(&Scr[0]);        // 32 cols x 256 tokens (T)
  const int wm4 = wave & 3, wn2 = wave >> 2;      // 4 x 2 wave grid (64x32)
  for (int ct = 0; ct < 64; ++ct) {
    const u16* SP = (ct < 32) ? V : U;
    float*     OP = (ct < 32) ? AttV : AttU;
    const int colbase = (ct & 31) << 5;
#pragma unroll
    for (int i = 0; i < 32; ++i) {                // load tile transposed
      int j = tid + (i << 8);
      int tok = j >> 5, col = j & 31;
      Vs[col][tok] = SP[(size_t)(base + tok) * HD + colbase + col];
    }
    __syncthreads();
    v8f acc = {};
#pragma unroll
    for (int ks = 0; ks < 8; ++ks) {
      const int k0 = ks << 5;
      const u16* arp = &AttS[wm4 * 16 + lr][0];
      v16bf af = frag16(arp + k0 + lh * 8, arp + k0 + lh * 8 + 16);
      const u16* brp = &Vs[wn2 * 16 + lr][0];
      v16bf bf = frag16(brp + k0 + lh * 8, brp + k0 + lh * 8 + 16);
      acc = __builtin_amdgcn_wmma_f32_16x16x32_bf16(
          false, af, false, bf, (short)0, acc, false, false);
    }
#pragma unroll
    for (int r = 0; r < 8; ++r) {
      int m = rbase + wm4 * 16 + r + (lh << 3);
      int n = colbase + wn2 * 16 + lr;
      OP[(size_t)m * HD + n] = acc[r];
    }
    __syncthreads();
  }
}

// ---------------------------------------------------------------------------
// Linear-attention KV summaries: KV[d,e] = inv_n * sum_t lin_k[t,d] * V[t,e]
// Split-K over tokens with f32 global atomics.
// ---------------------------------------------------------------------------
__global__ __launch_bounds__(256) void linkv_kernel(const u16* __restrict__ Lk,
    const u16* __restrict__ V, const u16* __restrict__ U,
    const float* __restrict__ inv_n_p, float* __restrict__ KV,
    float* __restrict__ KU) {
  const int bid = blockIdx.x;
  const int d  = bid & 127;
  const int ec = (bid >> 7) & 3;
  const int ts = bid >> 9;
  const int e  = ec * 256 + threadIdx.x;
  float accv = 0.f, accu = 0.f;
  const int t0 = ts * 2048, t1 = t0 + 2048;
  for (int t = t0; t < t1; ++t) {
    const float kd = bf2f(Lk[(size_t)t * QKD + d]);
    accv += kd * bf2f(V[(size_t)t * HD + e]);
    accu += kd * bf2f(U[(size_t)t * HD + e]);
  }
  const float inv_n = inv_n_p[0];
  atomicAdd(&KV[(size_t)d * HD + e], accv * inv_n);
  atomicAdd(&KU[(size_t)d * HD + e], accu * inv_n);
}

// ---------------------------------------------------------------------------
// Depthwise conv (K=17, 'same') fused variants.
// ---------------------------------------------------------------------------
__global__ __launch_bounds__(256) void dwconv_split_kernel(
    const u16* __restrict__ H, const float* __restrict__ W,
    u16* __restrict__ Vb, u16* __restrict__ Ub, int C) {
  const size_t idx = (size_t)blockIdx.x * 256 + threadIdx.x;
  const int c = (int)(idx % C);
  const int t = (int)(idx / C);
  float r = bf2f(H[idx]);
#pragma unroll
  for (int k = 0; k < KERN; ++k) {
    int tt = t + k - KPAD;
    if (tt >= 0 && tt < NT) r += bf2f(H[(size_t)tt * C + c]) * W[c * KERN + k];
  }
  const int half = C >> 1;
  if (c < half) Vb[(size_t)t * half + c] = f2bf(r);
  else          Ub[(size_t)t * half + (c - half)] = f2bf(r);
}

__global__ __launch_bounds__(256) void dwconv_qk_kernel(
    const u16* __restrict__ H, const float* __restrict__ W,
    const float* __restrict__ gam, const float* __restrict__ bet,
    u16* __restrict__ QQ, u16* __restrict__ LQ, u16* __restrict__ QK2,
    u16* __restrict__ LK) {
  const size_t idx = (size_t)blockIdx.x * 256 + threadIdx.x;
  const int c = (int)(idx % QKD);
  const int t = (int)(idx / QKD);
  float r = bf2f(H[idx]);
#pragma unroll
  for (int k = 0; k < KERN; ++k) {
    int tt = t + k - KPAD;
    if (tt >= 0 && tt < NT) r += bf2f(H[(size_t)tt * QKD + c]) * W[c * KERN + k];
  }
  QQ[idx]  = f2bf(r * gam[0 * QKD + c] + bet[0 * QKD + c]);
  LQ[idx]  = f2bf(r * gam[1 * QKD + c] + bet[1 * QKD + c]);
  QK2[idx] = f2bf(r * gam[2 * QKD + c] + bet[2 * QKD + c]);
  LK[idx]  = f2bf(r * gam[3 * QKD + c] + bet[3 * QKD + c]);
}

__global__ __launch_bounds__(256) void dwconv_final_kernel(
    const u16* __restrict__ G, const float* __restrict__ W,
    const float* __restrict__ X, float* __restrict__ Out) {
  const size_t idx = (size_t)blockIdx.x * 256 + threadIdx.x;
  const int c = (int)(idx % DIM);
  const int t = (int)(idx / DIM);
  float r = bf2f(G[idx]);
#pragma unroll
  for (int k = 0; k < KERN; ++k) {
    int tt = t + k - KPAD;
    if (tt >= 0 && tt < NT) r += bf2f(G[(size_t)tt * DIM + c]) * W[c * KERN + k];
  }
  Out[idx] = X[idx] + r;
}

// ---------------------------------------------------------------------------
// Elementwise helpers.
// ---------------------------------------------------------------------------
__global__ __launch_bounds__(256) void transpose_w_kernel(
    const float* __restrict__ W, u16* __restrict__ Wt, int K, int N) {
  const size_t idx = (size_t)blockIdx.x * 256 + threadIdx.x;
  const int n = (int)(idx % N), k = (int)(idx / N);
  Wt[(size_t)n * K + k] = f2bf(W[idx]);
}

__global__ __launch_bounds__(256) void zero_kernel(float* __restrict__ p, int n) {
  const int i = blockIdx.x * 256 + threadIdx.x;
  if (i < n) p[i] = 0.f;
}

__global__ __launch_bounds__(256) void gate_kernel(
    const float* __restrict__ AttV, const float* __restrict__ AttU,
    const u16* __restrict__ V, const u16* __restrict__ U,
    float* __restrict__ G) {
  const size_t idx = (size_t)blockIdx.x * 256 + threadIdx.x;
  const float v = bf2f(V[idx]);
  const float u = bf2f(U[idx]);
  G[idx] = AttU[idx] * v * sigmf(AttV[idx] * u);
}

// ---------------------------------------------------------------------------
// Launcher.
// ---------------------------------------------------------------------------
extern "C" void kernel_launch(void* const* d_in, const int* in_sizes, int n_in,
                              void* d_out, int out_size, void* d_ws,
                              size_t ws_size, hipStream_t stream) {
  (void)in_sizes; (void)n_in; (void)out_size; (void)ws_size;

  const float* x        = (const float*)d_in[0];
  const float* inv_n    = (const float*)d_in[1];
  const float* th_ln_g  = (const float*)d_in[2];
  const float* th_ln_b  = (const float*)d_in[3];
  const float* th_w     = (const float*)d_in[4];
  const float* th_b     = (const float*)d_in[5];
  const float* th_conv  = (const float*)d_in[6];
  const float* qk_ln_g  = (const float*)d_in[7];
  const float* qk_ln_b  = (const float*)d_in[8];
  const float* qk_w     = (const float*)d_in[9];
  const float* qk_b     = (const float*)d_in[10];
  const float* qk_conv  = (const float*)d_in[11];
  const float* os_gamma = (const float*)d_in[12];
  const float* os_beta  = (const float*)d_in[13];
  const float* out_ln_g = (const float*)d_in[14];
  const float* out_ln_b = (const float*)d_in[15];
  const float* out_w    = (const float*)d_in[16];
  const float* out_b    = (const float*)d_in[17];
  const float* out_conv = (const float*)d_in[18];

  char* ws = (char*)d_ws;
  size_t off = 0;
  auto alloc = [&](size_t bytes) -> char* {
    char* p = ws + off;
    off += (bytes + 255) & ~(size_t)255;
    return p;
  };
  u16*   xln_th = (u16*)alloc((size_t)NT * DIM * 2);
  u16*   xln_qk = (u16*)alloc((size_t)NT * DIM * 2);
  u16*   thWt   = (u16*)alloc((size_t)H2 * DIM * 2);
  u16*   qkWt   = (u16*)alloc((size_t)QKD * DIM * 2);
  u16*   outWt  = (u16*)alloc((size_t)DIM * HD * 2);
  u16*   hbuf   = (u16*)alloc((size_t)NT * H2 * 2);
  u16*   vbuf   = (u16*)alloc((size_t)NT * HD * 2);
  u16*   ubuf   = (u16*)alloc((size_t)NT * HD * 2);
  u16*   qkh    = (u16*)alloc((size_t)NT * QKD * 2);
  u16*   qq     = (u16*)alloc((size_t)NT * QKD * 2);
  u16*   lq     = (u16*)alloc((size_t)NT * QKD * 2);
  u16*   qk2    = (u16*)alloc((size_t)NT * QKD * 2);
  u16*   lk     = (u16*)alloc((size_t)NT * QKD * 2);
  float* attv   = (float*)alloc((size_t)NT * HD * 4);
  float* attu   = (float*)alloc((size_t)NT * HD * 4);
  float* kv     = (float*)alloc((size_t)QKD * HD * 4);
  float* ku     = (float*)alloc((size_t)QKD * HD * 4);
  u16*   kvT    = (u16*)alloc((size_t)QKD * HD * 2);
  u16*   kuT    = (u16*)alloc((size_t)QKD * HD * 2);
  // aliases (lifetimes do not overlap):
  float* gateb = (float*)hbuf;   // h dead after dwconv_split
  u16*   outln = (u16*)attv;     // attv dead after gate_kernel
  u16*   gbuf  = (u16*)attu;     // attu dead after gate_kernel

  // 1) weight transposes (f32 KxN -> bf16 NxK)
  transpose_w_kernel<<<(DIM * H2) / 256, 256, 0, stream>>>(th_w, thWt, DIM, H2);
  transpose_w_kernel<<<(DIM * QKD) / 256, 256, 0, stream>>>(qk_w, qkWt, DIM, QKD);
  transpose_w_kernel<<<(HD * DIM) / 256, 256, 0, stream>>>(out_w, outWt, HD, DIM);

  // 2) token-shift + LayerNorm (two parameter sets)
  ln_kernel<<<NT, 256, 0, stream>>>(x, th_ln_g, th_ln_b, xln_th, DIM, 1);
  ln_kernel<<<NT, 256, 0, stream>>>(x, qk_ln_g, qk_ln_b, xln_qk, DIM, 1);

  // 3) hidden / qk projections with fused SiLU
  gemm_kernel<<<dim3(H2 / BN, NT / BM), 256, 0, stream>>>(
      xln_th, thWt, th_b, hbuf, NT, H2, DIM, 1, 0);
  gemm_kernel<<<dim3(QKD / BN, NT / BM), 256, 0, stream>>>(
      xln_qk, qkWt, qk_b, qkh, NT, QKD, DIM, 1, 0);

  // 4) depthwise convs (+ residual) ; split v|u ; offset-scale qk into 4 heads
  dwconv_split_kernel<<<(NT * H2) / 256, 256, 0, stream>>>(
      hbuf, th_conv, vbuf, ubuf, H2);
  dwconv_qk_kernel<<<(NT * QKD) / 256, 256, 0, stream>>>(
      qkh, qk_conv, os_gamma, os_beta, qq, lq, qk2, lk);

  // 5) grouped quadratic attention -> attv/attu (overwrite)
  attn_kernel<<<(NT / GSZ) * 4, 256, 0, stream>>>(qq, qk2, vbuf, ubuf, attv, attu);

  // 6) global linear attention summaries (split-K + f32 atomics)
  zero_kernel<<<(QKD * HD) / 256, 256, 0, stream>>>(kv, QKD * HD);
  zero_kernel<<<(QKD * HD) / 256, 256, 0, stream>>>(ku, QKD * HD);
  linkv_kernel<<<128 * 4 * 8, 256, 0, stream>>>(lk, vbuf, ubuf, inv_n, kv, ku);
  transpose_w_kernel<<<(QKD * HD) / 256, 256, 0, stream>>>(kv, kvT, QKD, HD);
  transpose_w_kernel<<<(QKD * HD) / 256, 256, 0, stream>>>(ku, kuT, QKD, HD);

  // 7) lin_v / lin_u accumulated into attv / attu
  gemm_kernel<<<dim3(HD / BN, NT / BM), 256, 0, stream>>>(
      lq, kvT, nullptr, attv, NT, HD, QKD, 0, 1);
  gemm_kernel<<<dim3(HD / BN, NT / BM), 256, 0, stream>>>(
      lq, kuT, nullptr, attu, NT, HD, QKD, 0, 1);

  // 8) gated mixing -> f32, then output LayerNorm -> bf16
  gate_kernel<<<(NT * HD) / 256, 256, 0, stream>>>(attv, attu, vbuf, ubuf, gateb);
  ln_kernel<<<NT, 256, 0, stream>>>(gateb, out_ln_g, out_ln_b, outln, HD, 0);

  // 9) output projection (fused SiLU) + final depthwise conv + residual
  gemm_kernel<<<dim3(DIM / BN, NT / BM), 256, 0, stream>>>(
      outln, outWt, out_b, gbuf, NT, DIM, HD, 1, 0);
  dwconv_final_kernel<<<(NT * DIM) / 256, 256, 0, stream>>>(
      gbuf, out_conv, x, (float*)d_out);
}
